// SoftSplatBaseline_v3_39256001085883
// MI455X (gfx1250) — matrix operand order repaired and checked by reading, not compiled, not run on hardware
//
#include <hip/hip_runtime.h>
#include <hip/hip_bf16.h>
#include <math.h>

typedef __attribute__((ext_vector_type(16))) _Float16 v16h;
typedef __attribute__((ext_vector_type(8)))  _Float16 v8h;
typedef __attribute__((ext_vector_type(8)))  float    v8f;

#define SS_EPS 1e-7f
#define BPX    32   // pixels per block tile (two 16-wide WMMA N tiles)
#define LDSROW 40   // halves per pixel row: 32 data + 8 pad (80B stride, 16B aligned frags)

// ---------------------------------------------------------------------------
// Weight pack: OIHW f32 -> [Cout][Kpad] f16, zero-padded K (K = cin*9 + ky*3 + kx)
// ---------------------------------------------------------------------------
__global__ void pack_weights_kernel(const float* __restrict__ w, _Float16* __restrict__ dst,
                                    int Cout, int Kmax, int Kpad) {
  int i = blockIdx.x * blockDim.x + threadIdx.x;
  int n = Cout * Kpad;
  if (i >= n) return;
  int co = i / Kpad, k = i - co * Kpad;
  dst[i] = (k < Kmax) ? (_Float16)w[(size_t)co * Kmax + k] : (_Float16)0.f;
}

// ---------------------------------------------------------------------------
// Implicit-GEMM 3x3 conv (pad=1) with fused bias + PReLU.
// Block tile: 32 output pixels x Cout. Each wave owns 16 Cout rows and both
// 16-pixel N tiles: one A fragment feeds two V_WMMA_F32_16X16X32_F16 per
// K-chunk (two independent accumulators).
// im2col staging: one LDS row (32 pixels of one k) per wave iteration;
// wave id is readfirstlane'd so the k-decode, row base address and bounds
// checks are SALU and the load is a coalesced saddr+lane access.
// Fragment layouts per CDNA5 ISA 7.12.2.
// ---------------------------------------------------------------------------
__global__ void __launch_bounds__(192)
conv3x3_wmma_kernel(const float* __restrict__ in, const _Float16* __restrict__ wq,
                    const float* __restrict__ bias, const float* __restrict__ prelu_a, int aIdx,
                    float* __restrict__ out,
                    int Cin, int Cout, int Hin, int Win, int Hout, int Wout,
                    int stride, int Kpad) {
  __shared__ _Float16 ldsB[BPX * LDSROW];   // [pixel][k(+pad)] f16

  const int lane   = threadIdx.x & 31;
  const int wave   = __builtin_amdgcn_readfirstlane(threadIdx.x >> 5);   // wave-uniform -> SGPR
  const int nWaves = __builtin_amdgcn_readfirstlane((int)blockDim.x >> 5);
  const int wtiles = Wout >> 5;             // 32-pixel tiles per row

  int tile = blockIdx.x;
  const int x0 = (tile % wtiles) << 5; tile /= wtiles;
  const int oy = tile % Hout;
  const int b  = tile / Hout;

  const int Kmax = Cin * 9;
  const float alpha = prelu_a[aIdx];

  const int px = lane & 15;     // pixel column within N tile (B/C/D col)
  const int hi = lane >> 4;     // lane half select
  const int coutA = (wave << 4) + (lane & 15);          // A row (M)
  const _Float16* wrow = wq + (size_t)coutA * Kpad;

  const int ixBase = (x0 + lane) * stride - 1;          // per-lane x of kx=0 tap
  const int iyBase = oy * stride - 1;                   // scalar y of ky=0 tap

  v8f acc0 = {}, acc1 = {};

  for (int kb = 0; kb < Kpad; kb += 32) {
    // ---- cooperative im2col stage: one k-row (32 pixels) per wave iter ----
    for (int kl = wave; kl < 32; kl += nWaves) {
      int kk = kb + kl;                     // scalar (kl scalar)
      float v = 0.f;
      if (kk < Kmax) {                      // uniform branch (only layer-1 pads)
        int cin = kk / 9;                   // SALU decode, once per 32 elements
        int r   = kk - cin * 9;
        int ky  = r / 3;
        int kx  = r - ky * 3;
        int iy  = iyBase + ky;              // scalar
        int ix  = ixBase + kx;              // per-lane
        if ((unsigned)iy < (unsigned)Hin && (unsigned)ix < (unsigned)Win)
          v = in[(((size_t)b * Cin + cin) * Hin + iy) * Win + ix];
      }
      ldsB[lane * LDSROW + kl] = (_Float16)v;
    }
    __syncthreads();

    // ---- fragments ----
    // A: lane(l) row M=l&15; halves e0..7 -> K = 8*hi + e; e8..15 -> K = 16 + 8*hi + (e-8)
    union { v16h v; v8h h[2]; } afrag, bfrag0, bfrag1;
    const _Float16* ap = wrow + kb + (hi << 3);
    afrag.h[0] = *(const v8h*)(ap);
    afrag.h[1] = *(const v8h*)(ap + 16);
    // B: lane(l) col N=l&15; halves e0..7 -> K = 8*hi + e; e8..15 -> K = 16 + 8*hi + (e-8)
    const _Float16* bp0 = ldsB + px * LDSROW + (hi << 3);
    bfrag0.h[0] = *(const v8h*)(bp0);
    bfrag0.h[1] = *(const v8h*)(bp0 + 16);
    const _Float16* bp1 = ldsB + (px + 16) * LDSROW + (hi << 3);
    bfrag1.h[0] = *(const v8h*)(bp1);
    bfrag1.h[1] = *(const v8h*)(bp1 + 16);

    acc0 = __builtin_amdgcn_wmma_f32_16x16x32_f16(false, afrag.v, false, bfrag0.v,
                                                  (short)0, acc0, false, false);
    acc1 = __builtin_amdgcn_wmma_f32_16x16x32_f16(false, afrag.v, false, bfrag1.v,
                                                  (short)0, acc1, false, false);
    __syncthreads();
  }

  // ---- epilogue: C/D layout (M = r + 8*hi, N = px) -> bias + PReLU -> store ----
  const int nx0 = x0 + px;
  const int nx1 = x0 + 16 + px;
  #pragma unroll
  for (int r = 0; r < 8; ++r) {
    int m  = r + (hi << 3);
    int co = (wave << 4) + m;
    float bv = bias[co];
    size_t rowOff = (((size_t)b * Cout + co) * Hout + oy) * Wout;
    float v0 = acc0[r] + bv;
    v0 = (v0 >= 0.f) ? v0 : alpha * v0;
    out[rowOff + nx0] = v0;
    float v1 = acc1[r] + bv;
    v1 = (v1 >= 0.f) ? v1 : alpha * v1;
    out[rowOff + nx1] = v1;
  }
}

// ---------------------------------------------------------------------------
// z = param_scale * mean_c | (2*i0-1) - backwarp(2*i1-1, flow) |   (3 channels)
// ---------------------------------------------------------------------------
__global__ void zerr_kernel(const float* __restrict__ i0, const float* __restrict__ i1,
                            const float* __restrict__ flow, const float* __restrict__ pscale,
                            float* __restrict__ z, int Hh, int Ww, int Bn) {
  int idx = blockIdx.x * blockDim.x + threadIdx.x;
  int hw = Hh * Ww;
  if (idx >= Bn * hw) return;
  int p = idx % hw, b = idx / hw;
  int x = p % Ww,  y = p / Ww;
  const float* fb = flow + (size_t)b * 2 * hw;
  float px = fminf(fmaxf((float)x + fb[p],      0.f), (float)(Ww - 1));
  float py = fminf(fmaxf((float)y + fb[p + hw], 0.f), (float)(Hh - 1));
  float x0f = floorf(px), y0f = floorf(py);
  int x0 = (int)x0f, y0 = (int)y0f;
  float wx = px - x0f, wy = py - y0f;
  int x1 = min(x0 + 1, Ww - 1), y1 = min(y0 + 1, Hh - 1);
  float w00 = (1.f - wx) * (1.f - wy), w10 = wx * (1.f - wy);
  float w01 = (1.f - wx) * wy,         w11 = wx * wy;
  float err = 0.f;
  for (int c = 0; c < 3; ++c) {
    const float* s = i1 + ((size_t)b * 3 + c) * hw;
    float g = s[(size_t)y0 * Ww + x0] * w00 + s[(size_t)y0 * Ww + x1] * w10
            + s[(size_t)y1 * Ww + x0] * w01 + s[(size_t)y1 * Ww + x1] * w11;
    float p1 = 2.f * g - 1.f;
    float p0 = 2.f * i0[((size_t)b * 3 + c) * hw + p] - 1.f;
    err += fabsf(p0 - p1);
  }
  z[idx] = pscale[0] * (err * (1.f / 3.f));
}

// ---------------------------------------------------------------------------
// Bilinear resize, half-pixel coords, clamped (matches reference resize_bl)
// ---------------------------------------------------------------------------
__global__ void resize_bl_kernel(const float* __restrict__ in, float* __restrict__ out,
                                 int C, int Hin, int Win, int Hout, int Wout, int Bn) {
  int idx = blockIdx.x * blockDim.x + threadIdx.x;
  int total = Bn * C * Hout * Wout;
  if (idx >= total) return;
  int x = idx % Wout; int t = idx / Wout;
  int y = t % Hout;   t /= Hout;
  int c = t % C;      int b = t / C;
  float sy = (float)Hin / (float)Hout, sx = (float)Win / (float)Wout;
  float py = fminf(fmaxf(((float)y + 0.5f) * sy - 0.5f, 0.f), (float)(Hin - 1));
  float px = fminf(fmaxf(((float)x + 0.5f) * sx - 0.5f, 0.f), (float)(Win - 1));
  float y0f = floorf(py), x0f = floorf(px);
  int y0 = (int)y0f, x0 = (int)x0f;
  float wy = py - y0f, wx = px - x0f;
  int y1 = min(y0 + 1, Hin - 1), x1 = min(x0 + 1, Win - 1);
  const float* s = in + ((size_t)b * C + c) * Hin * Win;
  float r0 = s[(size_t)y0 * Win + x0] * (1.f - wx) + s[(size_t)y0 * Win + x1] * wx;
  float r1 = s[(size_t)y1 * Win + x0] * (1.f - wx) + s[(size_t)y1 * Win + x1] * wx;
  out[idx] = r0 * (1.f - wy) + r1 * wy;
}

__global__ void zero_kernel(float* __restrict__ p, int n) {
  int i = blockIdx.x * blockDim.x + threadIdx.x;
  if (i < n) p[i] = 0.f;
}

// ---------------------------------------------------------------------------
// softsplat scatter: f = tt*flow; vals = {-fx*ez, -fy*ez, ez}; 4-corner
// bilinear scatter-add into acc[b][3][h][w] with global_atomic_add_f32
// ---------------------------------------------------------------------------
__global__ void splat_kernel(const float* __restrict__ flow, const float* __restrict__ z,
                             const float* __restrict__ tptr, int dir,
                             float* __restrict__ acc, int Hh, int Ww, int Bn) {
  int idx = blockIdx.x * blockDim.x + threadIdx.x;
  int hw = Hh * Ww;
  if (idx >= Bn * hw) return;
  int p = idx % hw, b = idx / hw;
  int x = p % Ww,  y = p / Ww;
  float tt = tptr[b];
  if (dir) tt = 1.f - tt;
  const float* fb = flow + (size_t)b * 2 * hw;
  float fx = tt * fb[p], fy = tt * fb[p + hw];
  float ez = expf(z[idx]);
  float v0 = -fx * ez, v1 = -fy * ez;
  float gx = (float)x + fx, gy = (float)y + fy;
  float x0f = floorf(gx), y0f = floorf(gy);
  int x0 = (int)x0f, y0 = (int)y0f;
  float wx1 = gx - x0f, wy1 = gy - y0f;
  float wx0 = 1.f - wx1, wy0 = 1.f - wy1;
  int   cxs[4] = { x0, x0 + 1, x0, x0 + 1 };
  int   cys[4] = { y0, y0, y0 + 1, y0 + 1 };
  float cw [4] = { wx0 * wy0, wx1 * wy0, wx0 * wy1, wx1 * wy1 };
  float* ab = acc + (size_t)b * 3 * hw;
  #pragma unroll
  for (int i = 0; i < 4; ++i) {
    int cx = cxs[i], cy = cys[i];
    if (cx >= 0 && cx < Ww && cy >= 0 && cy < Hh) {
      int q = cy * Ww + cx;
      atomicAdd(&ab[q],            v0 * cw[i]);
      atomicAdd(&ab[hw + q],       v1 * cw[i]);
      atomicAdd(&ab[2 * hw + q],   ez * cw[i]);
    }
  }
}

__global__ void splat_norm_kernel(const float* __restrict__ acc, float* __restrict__ flowT,
                                  int Hh, int Ww, int Bn) {
  int idx = blockIdx.x * blockDim.x + threadIdx.x;
  int hw = Hh * Ww;
  if (idx >= Bn * hw) return;
  int p = idx % hw, b = idx / hw;
  const float* ab = acc + (size_t)b * 3 * hw;
  float d = ab[2 * hw + p] + SS_EPS;
  flowT[(size_t)b * 2 * hw + p]      = ab[p] / d;
  flowT[(size_t)b * 2 * hw + hw + p] = ab[hw + p] / d;
}

// ---------------------------------------------------------------------------
// backwarp src (C channels) with flowT, write into concatenated output at
// channel offset c0 of a totC-channel tensor
// ---------------------------------------------------------------------------
__global__ void backwarp_out_kernel(const float* __restrict__ src, const float* __restrict__ flow,
                                    float* __restrict__ dst, int C, int Hh, int Ww,
                                    int totC, int c0, int Bn) {
  int idx = blockIdx.x * blockDim.x + threadIdx.x;
  int hw = Hh * Ww;
  if (idx >= Bn * hw) return;
  int p = idx % hw, b = idx / hw;
  int x = p % Ww,  y = p / Ww;
  const float* fb = flow + (size_t)b * 2 * hw;
  float px = fminf(fmaxf((float)x + fb[p],      0.f), (float)(Ww - 1));
  float py = fminf(fmaxf((float)y + fb[p + hw], 0.f), (float)(Hh - 1));
  float x0f = floorf(px), y0f = floorf(py);
  int x0 = (int)x0f, y0 = (int)y0f;
  float wx = px - x0f, wy = py - y0f;
  int x1 = min(x0 + 1, Ww - 1), y1 = min(y0 + 1, Hh - 1);
  float w00 = (1.f - wx) * (1.f - wy), w10 = wx * (1.f - wy);
  float w01 = (1.f - wx) * wy,         w11 = wx * wy;
  for (int c = 0; c < C; ++c) {
    const float* s = src + ((size_t)b * C + c) * hw;
    float g = s[(size_t)y0 * Ww + x0] * w00 + s[(size_t)y0 * Ww + x1] * w10
            + s[(size_t)y1 * Ww + x0] * w01 + s[(size_t)y1 * Ww + x1] * w11;
    dst[(((size_t)b * totC + c0 + c) * Hh + y) * Ww + x] = g;
  }
}

// ---------------------------------------------------------------------------
extern "C" void kernel_launch(void* const* d_in, const int* in_sizes, int n_in,
                              void* d_out, int out_size, void* d_ws, size_t ws_size,
                              hipStream_t stream) {
  (void)in_sizes; (void)n_in; (void)out_size; (void)ws_size;
  const int B = 2, H = 512, W = 512;
  const size_t HW = (size_t)H * W;

  const float* input0 = (const float*)d_in[0];
  const float* input1 = (const float*)d_in[1];
  const float* tgt    = (const float*)d_in[2];
  const float* flows[2][3] = {
    { (const float*)d_in[3], (const float*)d_in[4], (const float*)d_in[5] },
    { (const float*)d_in[6], (const float*)d_in[7], (const float*)d_in[8] } };
  const float* wgt[6]; const float* bia[6];
  for (int i = 0; i < 6; ++i) { wgt[i] = (const float*)d_in[9 + 2*i]; bia[i] = (const float*)d_in[10 + 2*i]; }
  const float* prelu_a = (const float*)d_in[21];
  const float* pscale  = (const float*)d_in[22];

  struct LCfg { int Cin, Cout, Hin, Win, Hout, Wout, stride, Kpad; };
  const LCfg L[6] = {
    {  3, 32, 512, 512, 512, 512, 1,  32 },
    { 32, 32, 512, 512, 512, 512, 1, 288 },
    { 32, 64, 512, 512, 256, 256, 2, 288 },
    { 64, 64, 256, 256, 256, 256, 1, 576 },
    { 64, 96, 256, 256, 128, 128, 2, 576 },
    { 96, 96, 128, 128, 128, 128, 1, 864 },
  };

  // ---- deterministic workspace carve-up ----
  char* ws = (char*)d_ws;
  size_t off = 0;
  auto carve = [&](size_t bytes) -> char* {
    char* p = ws + off;
    off = (off + bytes + 255) & ~(size_t)255;
    return p;
  };
  _Float16* wq[6];
  for (int i = 0; i < 6; ++i) wq[i] = (_Float16*)carve((size_t)L[i].Cout * L[i].Kpad * sizeof(_Float16));
  float* tmp = (float*)carve((size_t)B * 32 * HW * sizeof(float));
  float *fea1[2], *fea2[2], *fea3[2];
  for (int i = 0; i < 2; ++i) fea1[i] = (float*)carve((size_t)B * 32 * HW      * sizeof(float));
  for (int i = 0; i < 2; ++i) fea2[i] = (float*)carve((size_t)B * 64 * (HW/4)  * sizeof(float));
  for (int i = 0; i < 2; ++i) fea3[i] = (float*)carve((size_t)B * 96 * (HW/16) * sizeof(float));
  float* zbuf  = (float*)carve((size_t)B * HW       * sizeof(float));
  float* zs    = (float*)carve((size_t)B * (HW/4)   * sizeof(float));
  float* accb  = (float*)carve((size_t)B * 3 * HW   * sizeof(float));
  float* flowT = (float*)carve((size_t)B * 2 * HW   * sizeof(float));

  // ---- weight pack (f32 OIHW -> f16 [Cout][Kpad]) ----
  for (int i = 0; i < 6; ++i) {
    int n = L[i].Cout * L[i].Kpad;
    pack_weights_kernel<<<(n + 255) / 256, 256, 0, stream>>>(wgt[i], wq[i], L[i].Cout, L[i].Cin * 9, L[i].Kpad);
  }

  // ---- feature extraction (WMMA convs), both images ----
  const float* imgsIn[2] = { input0, input1 };
  float* feaAll[2][3];
  for (int im = 0; im < 2; ++im) {
    const float* cur = imgsIn[im];
    float* outs[6] = { tmp, fea1[im], tmp, fea2[im], tmp, fea3[im] };
    for (int li = 0; li < 6; ++li) {
      const LCfg& c = L[li];
      dim3 grid((unsigned)(B * c.Hout * (c.Wout / BPX)));
      dim3 block((unsigned)(32 * (c.Cout / 16)));
      conv3x3_wmma_kernel<<<grid, block, 0, stream>>>(cur, wq[li], bia[li], prelu_a, li,
          outs[li], c.Cin, c.Cout, c.Hin, c.Win, c.Hout, c.Wout, c.stride, c.Kpad);
      cur = outs[li];
    }
    feaAll[im][0] = fea1[im]; feaAll[im][1] = fea2[im]; feaAll[im][2] = fea3[im];
  }

  // ---- output bases (concatenation layout) ----
  float* outF  = (float*)d_out;
  float* outL0 = outF;                                   // (B,70,512,512)
  float* outL1 = outF + (size_t)B * 70 * HW;             // (B,128,256,256)
  float* outL2 = outL1 + (size_t)B * 128 * (HW / 4);     // (B,192,128,128)

  // ---- per-direction splat + warp ----
  for (int d = 0; d < 2; ++d) {
    const float* i0 = (d == 0) ? input0 : input1;
    const float* i1 = (d == 0) ? input1 : input0;
    {
      int n = B * H * W;
      zerr_kernel<<<(n + 255) / 256, 256, 0, stream>>>(i0, i1, flows[d][0], pscale, zbuf, H, W, B);
    }
    for (int s = 0; s < 3; ++s) {
      int hh = H >> s, ww = W >> s, hw = hh * ww;
      const float* zcur = zbuf;
      if (s > 0) {
        int n = B * hw;
        resize_bl_kernel<<<(n + 255) / 256, 256, 0, stream>>>(zbuf, zs, 1, H, W, hh, ww, B);
        zcur = zs;
      }
      {
        int na = B * 3 * hw;
        zero_kernel<<<(na + 255) / 256, 256, 0, stream>>>(accb, na);
      }
      int n = B * hw;
      splat_kernel<<<(n + 255) / 256, 256, 0, stream>>>(flows[d][s], zcur, tgt, d, accb, hh, ww, B);
      splat_norm_kernel<<<(n + 255) / 256, 256, 0, stream>>>(accb, flowT, hh, ww, B);
      if (s == 0) {
        backwarp_out_kernel<<<(n + 255) / 256, 256, 0, stream>>>(i0, flowT, outL0, 3,  hh, ww, 70, d ? 35 : 0, B);
        backwarp_out_kernel<<<(n + 255) / 256, 256, 0, stream>>>(feaAll[d][0], flowT, outL0, 32, hh, ww, 70, d ? 38 : 3, B);
      } else if (s == 1) {
        backwarp_out_kernel<<<(n + 255) / 256, 256, 0, stream>>>(feaAll[d][1], flowT, outL1, 64, hh, ww, 128, d * 64, B);
      } else {
        backwarp_out_kernel<<<(n + 255) / 256, 256, 0, stream>>>(feaAll[d][2], flowT, outL2, 96, hh, ww, 192, d * 96, B);
      }
    }
  }
}
